// EdgeWeightNormalizedMaxKSAGEConv_85366769975857
// MI455X (gfx1250) — matrix operand order repaired and checked by reading, not compiled, not run on hardware
//
#include <hip/hip_runtime.h>

#define N_NODES 50000
#define N_EDGES 800000
#define FEAT 128
#define KTOP 32
#define RT_TOTAL (N_NODES / 16)   // 3125 row tiles, divides exactly

typedef __attribute__((ext_vector_type(16))) __bf16 v16bf;
typedef __attribute__((ext_vector_type(8)))  float  v8f;

// ---------------- zero agg + deg ----------------
__global__ void k_zero(float* agg, float* deg) {
    int i = blockIdx.x * blockDim.x + threadIdx.x;
    if (i < N_NODES * FEAT) agg[i] = 0.0f;
    if (i < N_NODES)        deg[i] = 0.0f;
}

// ---- per-row dedup mask: bit k set iff k is the LAST occurrence of its index
// (matches .at[rows, idx].set(vals) last-wins semantics) ----
__global__ void k_mask(const int* __restrict__ topk_idx, unsigned* __restrict__ mask) {
    int n = blockIdx.x * blockDim.x + threadIdx.x;
    if (n >= N_NODES) return;
    const int* p = topk_idx + n * KTOP;
    unsigned long long lo = 0ull, hi = 0ull;
    unsigned m = 0u;
    for (int k = KTOP - 1; k >= 0; --k) {
        int idx = p[k] & 127;
        bool seen;
        if (idx < 64) { seen = (lo >> idx) & 1ull; lo |= 1ull << idx; }
        else          { int b = idx - 64; seen = (hi >> b) & 1ull; hi |= 1ull << b; }
        if (!seen) m |= 1u << k;
    }
    mask[n] = m;
}

// ---------------- in-degree via float atomics ----------------
__global__ void k_degree(const int* __restrict__ dst, float* __restrict__ deg) {
    int e = blockIdx.x * blockDim.x + threadIdx.x;
    if (e < N_EDGES) atomicAdd(&deg[dst[e]], 1.0f);
}

// ---------------- sparse SpMM: one wave32 per edge, lane = k ----------------
__global__ void k_spmm(const int* __restrict__ src, const int* __restrict__ dst,
                       const float* __restrict__ topk_val, const int* __restrict__ topk_idx,
                       const unsigned* __restrict__ mask, const float* __restrict__ deg,
                       float* __restrict__ agg) {
    int gid  = blockIdx.x * blockDim.x + threadIdx.x;
    int e    = gid >> 5;
    int lane = threadIdx.x & 31;
    if (e >= N_EDGES) return;
    int s = src[e], d = dst[e];
    float w = 1.0f / fmaxf(deg[d], 1.0f);
    unsigned m = mask[s];
    if ((m >> lane) & 1u) {
        int   idx = topk_idx[s * KTOP + lane] & 127;
        float v   = topk_val[s * KTOP + lane];
        atomicAdd(&agg[d * FEAT + idx], w * v);
    }
}

// ---- pre-swizzle W_self/W_neigh into bf16 hi/lo fragments in the WMMA
// B-matrix lane layout: wfrag[m][t][j][lane][i], m: 0=Whi_self 1=Wlo_self
// 2=Whi_neigh 3=Wlo_neigh; B element (v=i/2, s=i%2, h=lane/16): v<4 -> k=2v+s+8h, v>=4 -> k=2v+s+8+8h ----
__global__ void k_wfrag(const float* __restrict__ Wself, const float* __restrict__ Wneigh,
                        __bf16* __restrict__ wfrag) {
    int d = blockIdx.x * blockDim.x + threadIdx.x;   // 2 * 4 * 8 * 32 * 16 = 32768
    if (d >= 32768) return;
    int i    = d & 15;
    int lane = (d >> 4) & 31;
    int j    = (d >> 9) & 7;
    int t    = (d >> 12) & 3;
    int mm   = (d >> 14) & 1;
    int v = i >> 1, s = i & 1, h = lane >> 4;
    int k = 32 * t + ((v < 4) ? (2 * v + s + 8 * h) : (2 * v + s + 8 + 8 * h));
    int n = 16 * j + (lane & 15);
    const float* W = mm ? Wneigh : Wself;
    float  x  = W[k * FEAT + n];
    __bf16 hi = (__bf16)x;
    __bf16 lo = (__bf16)(x - (float)hi);
    int base = (((2 * mm) * 4 + t) * 8 + j) * 512 + lane * 16 + i;  // m stride = 4*8*512 = 16384
    wfrag[base]         = hi;
    wfrag[base + 16384] = lo;
}

// ---------------- fused GEMM: out = feat@Wself + b + agg@Wneigh ----------------
__global__ void __launch_bounds__(256)
k_gemm(const float* __restrict__ feat, const float* __restrict__ agg,
       const __bf16* __restrict__ wfrag, const float* __restrict__ b_self,
       float* __restrict__ out) {
    int lane = threadIdx.x & 31;
    int rt   = blockIdx.x * 8 + (threadIdx.x >> 5);   // one 16-row strip per wave
    if (rt >= RT_TOTAL) return;                        // whole-wave exit: EXEC stays all-1s
    int h = lane >> 4, ln = lane & 15;
    int row = rt * 16 + ln;

    v8f acc[8];
#pragma unroll
    for (int j = 0; j < 8; ++j) {
        float bv = b_self[16 * j + ln];
#pragma unroll
        for (int p = 0; p < 8; ++p) acc[j][p] = bv;
    }

    const float* frow = feat + row * FEAT;
    const float* arow = agg  + row * FEAT;

    for (int t = 0; t < 4; ++t) {
        // A fragments (16-bit A 16x32 layout): v<4 -> k=2v+s+8h, v>=4 -> k=2v+s+8+8h
        v16bf fh, fl, gh, gl;
#pragma unroll
        for (int v = 0; v < 8; ++v) {
            int kb = 32 * t + ((v < 4) ? (2 * v + 8 * h) : (2 * v + 8 + 8 * h));
            float2 f2 = *(const float2*)(frow + kb);
            float2 a2 = *(const float2*)(arow + kb);
            __bf16 x0 = (__bf16)f2.x; fh[2*v]   = x0; fl[2*v]   = (__bf16)(f2.x - (float)x0);
            __bf16 x1 = (__bf16)f2.y; fh[2*v+1] = x1; fl[2*v+1] = (__bf16)(f2.y - (float)x1);
            __bf16 y0 = (__bf16)a2.x; gh[2*v]   = y0; gl[2*v]   = (__bf16)(a2.x - (float)y0);
            __bf16 y1 = (__bf16)a2.y; gh[2*v+1] = y1; gl[2*v+1] = (__bf16)(a2.y - (float)y1);
        }
#pragma unroll
        for (int j = 0; j < 8; ++j) {
            const __bf16* bp = wfrag + (t * 8 + j) * 512 + lane * 16;
            v16bf bsh = *(const v16bf*)(bp);              // W_self  hi
            v16bf bsl = *(const v16bf*)(bp + 16384);      // W_self  lo
            v16bf bnh = *(const v16bf*)(bp + 32768);      // W_neigh hi
            v16bf bnl = *(const v16bf*)(bp + 49152);      // W_neigh lo
            v8f c = acc[j];
            // split-bf16 near-f32 GEMM: Ah*Bh + Al*Bh + Ah*Bl, f32 accumulate
            c = __builtin_amdgcn_wmma_f32_16x16x32_bf16(false, fh, false, bsh, (short)0, c, false, false);
            c = __builtin_amdgcn_wmma_f32_16x16x32_bf16(false, fl, false, bsh, (short)0, c, false, false);
            c = __builtin_amdgcn_wmma_f32_16x16x32_bf16(false, fh, false, bsl, (short)0, c, false, false);
            c = __builtin_amdgcn_wmma_f32_16x16x32_bf16(false, gh, false, bnh, (short)0, c, false, false);
            c = __builtin_amdgcn_wmma_f32_16x16x32_bf16(false, gl, false, bnh, (short)0, c, false, false);
            c = __builtin_amdgcn_wmma_f32_16x16x32_bf16(false, gh, false, bnl, (short)0, c, false, false);
            acc[j] = c;
        }
    }

    // C/D layout: VGPR p -> M = p + 8h, N = 16j + ln
#pragma unroll
    for (int j = 0; j < 8; ++j) {
#pragma unroll
        for (int p = 0; p < 8; ++p) {
            out[(rt * 16 + p + 8 * h) * FEAT + 16 * j + ln] = acc[j][p];
        }
    }
}

extern "C" void kernel_launch(void* const* d_in, const int* in_sizes, int n_in,
                              void* d_out, int out_size, void* d_ws, size_t ws_size,
                              hipStream_t stream) {
    const float*  feat     = (const float*)d_in[0];
    const float*  topk_val = (const float*)d_in[1];
    const int*    topk_idx = (const int*)d_in[2];
    const int*    src      = (const int*)d_in[3];
    const int*    dst      = (const int*)d_in[4];
    const float*  Wself    = (const float*)d_in[5];
    const float*  bself    = (const float*)d_in[6];
    const float*  Wneigh   = (const float*)d_in[7];
    float*        out      = (float*)d_out;

    char* ws = (char*)d_ws;
    float*    agg   = (float*)(ws);                   // N*128*4  = 25,600,000 B
    float*    deg   = (float*)(ws + 25600000);        // N*4      =    200,000 B
    unsigned* mask  = (unsigned*)(ws + 25800000);     // N*4      =    200,000 B
    __bf16*   wfrag = (__bf16*)(ws + 26000000);       // 4*16384*2 =   131,072 B (32B aligned)

    k_zero  <<<(N_NODES * FEAT + 255) / 256, 256, 0, stream>>>(agg, deg);
    k_mask  <<<(N_NODES + 255) / 256,        256, 0, stream>>>(topk_idx, mask);
    k_degree<<<(N_EDGES + 255) / 256,        256, 0, stream>>>(dst, deg);
    k_wfrag <<<32768 / 256,                  256, 0, stream>>>(Wself, Wneigh, wfrag);
    k_spmm  <<<(N_EDGES * 32) / 256,         256, 0, stream>>>(src, dst, topk_val, topk_idx, mask, deg, agg);
    k_gemm  <<<(RT_TOTAL + 7) / 8,           256, 0, stream>>>(feat, agg, wfrag, bself, out);
}
